// EnhancedMambaOnly_47605417509005
// MI455X (gfx1250) — compile-verified, matrix operand order checked by reading, not run on hardware
//
#include <hip/hip_runtime.h>
#include <cstdint>
#include <cstddef>

// ---------------------------------------------------------------------------
// Graph-Mamba (bidirectional) for MI455X / gfx1250, wave32 + WMMA bf16.
//   dense = x reshaped (64 x 1024 x 256)   (perm is the identity; edges unused)
//   f = mamba_fwd(dense); b = flip(mamba_bwd(flip(dense)))  [flips via indexing]
//   y = sigmoid([f|b] @ gW^T + gb) blended, LN1, FFN(relu), +x residual, LN2
// GEMMs: v_wmma_f32_16x16x32_bf16; weight tiles staged to LDS with
// global_load_async_to_lds_b128 (ASYNCcnt) and read back via ds_load_b128.
// ---------------------------------------------------------------------------

#define NTOK   65536
#define LSEQ   1024
#define NGRAPH 64
#define DMODEL 256
#define DIN    512
#define DSTATE 16
#define LN_EPS 1e-5f

typedef __bf16 bf16;
typedef __attribute__((ext_vector_type(16))) __bf16 v16bf;
typedef __attribute__((ext_vector_type(8)))  __bf16 v8bf;
typedef __attribute__((ext_vector_type(8)))  float  v8f;

enum { ACT_NONE = 0, ACT_SOFTPLUS = 1, ACT_SIGMOID = 2, ACT_RELU = 3 };

// -------------------------- WMMA fragment load -----------------------------
// 16-bit A/B 16x32 layout (ISA 7.12.2): lane half 0 holds K {0..7,16..23},
// half 1 holds K {8..15,24..31}; p must already include base + ko + half*8.
__device__ __forceinline__ v16bf load_frag(const bf16* p) {
  v8bf lo = *(const v8bf*)(p);
  v8bf hi = *(const v8bf*)(p + 16);
  v16bf r;
#pragma unroll
  for (int j = 0; j < 8; ++j) { r[j] = lo[j]; r[j + 8] = hi[j]; }
  return r;
}

// ------------------------------ GEMM kernel --------------------------------
// out[M,N] = act(A[M,K] @ W[N,K]^T + bias).  Block = 4 waves along M,
// each wave owns a 16 x (16*TN) tile.  The block's weight tile (16*TN x K)
// is async-copied global->LDS once, then the K-loop reads B frags from LDS
// while A frags stream from global with one-step software pipelining.
template <int K, int TN, int ACT, bool HAS_BIAS, bool OUT_BF16, bool DUAL>
__global__ void __launch_bounds__(128)
gemm_wmma(const bf16* __restrict__ A, const bf16* __restrict__ W,
          const float* __restrict__ bias, void* __restrict__ out, int N,
          bf16* __restrict__ out2, int ldo2, int coloff) {
  __shared__ __align__(16) bf16 sW[16 * TN * K];

  const int lane = threadIdx.x & 31;
  const int wave = threadIdx.x >> 5;
  const int half = lane >> 4;
  const int lm   = lane & 15;
  const int m0 = (blockIdx.x * 4 + wave) * 16;
  const int n0 = blockIdx.y * (16 * TN);

  // ---- async-stage the W tile into LDS (16B chunks, strided over block) ----
  {
    const uint32_t lds0 = (uint32_t)(uintptr_t)&sW[0];
    const uint64_t g0   = (uint64_t)(uintptr_t)(W + (size_t)n0 * K);
    constexpr int NCHUNK = (16 * TN * K * 2) / 16;
#pragma unroll 1
    for (int i = threadIdx.x; i < NCHUNK; i += 128) {
      uint32_t ldsa = lds0 + (uint32_t)i * 16u;
      uint64_t ga   = g0 + (uint64_t)i * 16u;
      asm volatile("global_load_async_to_lds_b128 %0, %1, off"
                   :: "v"(ldsa), "v"(ga) : "memory");
    }
    asm volatile("s_wait_asynccnt 0x0" ::: "memory");
  }
  __syncthreads();

  const bf16* arow = A + (size_t)(m0 + lm) * K + half * 8;
  const bf16* brow = &sW[0] + (size_t)lm * K + half * 8;

  v8f acc[TN];
#pragma unroll
  for (int t = 0; t < TN; ++t)
#pragma unroll
    for (int r = 0; r < 8; ++r) acc[t][r] = 0.0f;

  v16bf av = load_frag(arow);
#pragma unroll
  for (int ko = 0; ko < K; ko += 32) {
    v16bf avn = av;
    if (ko + 32 < K) avn = load_frag(arow + ko + 32);   // prefetch next step
#pragma unroll
    for (int t = 0; t < TN; ++t) {
      v16bf bv = load_frag(brow + (size_t)t * 16 * K + ko);   // ds_load_b128
      acc[t] = __builtin_amdgcn_wmma_f32_16x16x32_bf16(
          false, av, false, bv, (short)0, acc[t], false, false);
    }
    av = avn;
  }

#pragma unroll
  for (int t = 0; t < TN; ++t) {
    const int col = n0 + t * 16 + lm;
    const float bv = HAS_BIAS ? bias[col] : 0.0f;
#pragma unroll
    for (int r = 0; r < 8; ++r) {
      const int row = m0 + half * 8 + r;
      float v = acc[t][r] + bv;
      if (ACT == ACT_SOFTPLUS) v = (v > 20.0f) ? v : log1pf(__expf(v));
      else if (ACT == ACT_SIGMOID) v = 1.0f / (1.0f + __expf(-v));
      else if (ACT == ACT_RELU) v = v > 0.0f ? v : 0.0f;
      if (OUT_BF16) ((bf16*)out)[(size_t)row * N + col] = (bf16)v;
      else          ((float*)out)[(size_t)row * N + col] = v;
      if (DUAL) out2[(size_t)row * ldo2 + coloff + col] = (bf16)v;
    }
  }
}

// ------------------------------ small kernels ------------------------------
__global__ void __launch_bounds__(256)
cvt_f32_bf16(const float* __restrict__ in, bf16* __restrict__ out, int n) {
  int i = blockIdx.x * 256 + threadIdx.x;
  if (i < n) out[i] = (bf16)in[i];
}

// Wdt_eff[c,k] = sum_r Wdt[c,r] * Wx[r,k]   (folds dt projection into one GEMM)
__global__ void __launch_bounds__(256)
wdt_eff_kernel(const float* __restrict__ Wdt, const float* __restrict__ Wx,
               bf16* __restrict__ out) {
  int i = blockIdx.x * 256 + threadIdx.x;   // c*512 + k
  if (i >= DIN * DIN) return;
  int c = i >> 9, k = i & 511;
  float s = 0.0f;
#pragma unroll
  for (int r = 0; r < 16; ++r) s += Wdt[c * 16 + r] * Wx[r * DIN + k];
  out[i] = (bf16)s;
}

// Depthwise causal conv (width 4) + SiLU.  REV handles the bwd-flip by
// reversed-direction taps (flip->conv->flip == anticausal conv).
template <bool REV>
__global__ void __launch_bounds__(256)
conv_silu_kernel(const bf16* __restrict__ xz, const float* __restrict__ convw,
                 const float* __restrict__ convb, bf16* __restrict__ xc) {
  long long idx = (long long)blockIdx.x * 256 + threadIdx.x;  // token*512 + c
  if (idx >= (long long)NTOK * DIN) return;
  int c = (int)(idx & (DIN - 1));
  long long token = idx >> 9;
  int t = (int)(token & (LSEQ - 1));
  float s = convb[c];
#pragma unroll
  for (int k = 0; k < 4; ++k) {
    int tt = REV ? (t + 3 - k) : (t - 3 + k);
    if (tt >= 0 && tt < LSEQ) {
      long long tok2 = token + (tt - t);
      s += convw[c * 4 + k] * (float)xz[tok2 * (2 * DIN) + c];
    }
  }
  xc[idx] = (bf16)(s / (1.0f + __expf(-s)));
}

// Selective scan: one lane per (graph, channel); 16 states in registers;
// B/C (shared by all channels of a graph) staged through LDS each step.
// Fuses u*D skip and the silu(z) gate so yg is written once in bf16.
template <bool REV>
__global__ void __launch_bounds__(256)
scan_kernel(const bf16* __restrict__ xc, const bf16* __restrict__ dtb,
            const float* __restrict__ bc, const bf16* __restrict__ xz,
            const float* __restrict__ Alog, const float* __restrict__ Dp,
            bf16* __restrict__ yg) {
  const int g = blockIdx.x >> 1;
  const int c = ((blockIdx.x & 1) << 8) + threadIdx.x;
  float A[DSTATE];
#pragma unroll
  for (int s = 0; s < DSTATE; ++s) A[s] = -__expf(Alog[c * DSTATE + s]);
  const float Dc = Dp[c];
  float h[DSTATE];
#pragma unroll
  for (int s = 0; s < DSTATE; ++s) h[s] = 0.0f;

  __shared__ float sBC[32];
  for (int step = 0; step < LSEQ; ++step) {
    const int t = REV ? (LSEQ - 1 - step) : step;
    const size_t tok = (size_t)g * LSEQ + t;
    if (threadIdx.x < 32) sBC[threadIdx.x] = bc[tok * 32 + threadIdx.x];
    __syncthreads();
    const float dtv = (float)dtb[tok * DIN + c];
    const float u   = (float)xc[tok * DIN + c];
    const float du  = dtv * u;
    float acc = 0.0f;
#pragma unroll
    for (int s = 0; s < DSTATE; ++s) {
      float dA = __expf(dtv * A[s]);
      h[s] = dA * h[s] + du * sBC[s];
      acc += h[s] * sBC[16 + s];
    }
    const float zv  = (float)xz[tok * (2 * DIN) + DIN + c];
    const float sil = zv / (1.0f + __expf(-zv));
    yg[tok * DIN + c] = (bf16)((acc + u * Dc) * sil);
    __syncthreads();
  }
}

// LayerNorm over 256 features, one token per 256-thread block.
__device__ __forceinline__ float ln256(float y, float g, float b, float* red) {
  const int c = threadIdx.x;
  red[c] = y; __syncthreads();
  for (int s = 128; s > 0; s >>= 1) { if (c < s) red[c] += red[c + s]; __syncthreads(); }
  float mean = red[0] * (1.0f / 256.0f);
  __syncthreads();
  float d = y - mean; red[c] = d * d; __syncthreads();
  for (int s = 128; s > 0; s >>= 1) { if (c < s) red[c] += red[c + s]; __syncthreads(); }
  float var = red[0] * (1.0f / 256.0f);
  __syncthreads();
  return (y - mean) * rsqrtf(var + LN_EPS) * g + b;
}

__global__ void __launch_bounds__(256)
combine_ln1_kernel(const float* __restrict__ gate, const float* __restrict__ f,
                   const float* __restrict__ b, const float* __restrict__ g1,
                   const float* __restrict__ b1, bf16* __restrict__ gmb) {
  __shared__ float red[256];
  size_t i = (size_t)blockIdx.x * 256 + threadIdx.x;
  float gv = gate[i];
  float y = gv * f[i] + (1.0f - gv) * b[i];
  gmb[i] = (bf16)ln256(y, g1[threadIdx.x], b1[threadIdx.x], red);
}

__global__ void __launch_bounds__(256)
resid_ln2_kernel(const float* __restrict__ fe, const float* __restrict__ x,
                 const float* __restrict__ g2, const float* __restrict__ b2,
                 float* __restrict__ out) {
  __shared__ float red[256];
  size_t i = (size_t)blockIdx.x * 256 + threadIdx.x;
  float y = fe[i] + x[i];
  out[i] = ln256(y, g2[threadIdx.x], b2[threadIdx.x], red);
}

// ------------------------------- host side ---------------------------------
struct MambaParams {
  const float *Win, *convw, *convb, *Wx, *Wdt, *bdt, *Alog, *D, *Wout;
};

extern "C" void kernel_launch(void* const* d_in, const int* in_sizes, int n_in,
                              void* d_out, int out_size, void* d_ws, size_t ws_size,
                              hipStream_t stream) {
  (void)in_sizes; (void)n_in; (void)out_size; (void)ws_size;

  const float* x = (const float*)d_in[0];
  // d_in[1] edge_index, d_in[2] edge_attr, d_in[3] batch: unused by reference math.
  auto getm = [&](int base) {
    MambaParams m;
    m.Win  = (const float*)d_in[base + 0];
    m.convw= (const float*)d_in[base + 1];
    m.convb= (const float*)d_in[base + 2];
    m.Wx   = (const float*)d_in[base + 3];
    m.Wdt  = (const float*)d_in[base + 4];
    m.bdt  = (const float*)d_in[base + 5];
    m.Alog = (const float*)d_in[base + 6];
    m.D    = (const float*)d_in[base + 7];
    m.Wout = (const float*)d_in[base + 8];
    return m;
  };
  MambaParams mp[2] = { getm(4), getm(13) };
  const float* gate_W = (const float*)d_in[22];
  const float* gate_b = (const float*)d_in[23];
  const float* ln1_g  = (const float*)d_in[24];
  const float* ln1_b  = (const float*)d_in[25];
  const float* ln2_g  = (const float*)d_in[26];
  const float* ln2_b  = (const float*)d_in[27];
  const float* fe_W1  = (const float*)d_in[28];
  const float* fe_b1  = (const float*)d_in[29];
  const float* fe_W2  = (const float*)d_in[30];
  const float* fe_b2  = (const float*)d_in[31];
  float* outp = (float*)d_out;

  // ---- workspace bump allocator (~582 MB total) ----
  uint8_t* ws = (uint8_t*)d_ws;
  size_t off = 0;
  auto alloc = [&](size_t bytes) {
    size_t o = off; off += (bytes + 255) & ~(size_t)255; return o;
  };
  bf16*  xb    = (bf16*)(ws + alloc((size_t)NTOK * DMODEL * 2));
  bf16*  win16[2]  = { (bf16*)(ws + alloc(1024 * 256 * 2)),
                       (bf16*)(ws + alloc(1024 * 256 * 2)) };
  bf16*  wx16[2]   = { (bf16*)(ws + alloc(48 * 512 * 2)),
                       (bf16*)(ws + alloc(48 * 512 * 2)) };
  bf16*  wdte16[2] = { (bf16*)(ws + alloc(512 * 512 * 2)),
                       (bf16*)(ws + alloc(512 * 512 * 2)) };
  bf16*  wout16[2] = { (bf16*)(ws + alloc(256 * 512 * 2)),
                       (bf16*)(ws + alloc(256 * 512 * 2)) };
  bf16*  gw16  = (bf16*)(ws + alloc(256 * 512 * 2));
  bf16*  w116  = (bf16*)(ws + alloc(256 * 256 * 2));
  bf16*  w216  = (bf16*)(ws + alloc(256 * 256 * 2));
  float* f_out = (float*)(ws + alloc((size_t)NTOK * DMODEL * 4));
  float* b_out = (float*)(ws + alloc((size_t)NTOK * DMODEL * 4));
  bf16*  fb16  = (bf16*)(ws + alloc((size_t)NTOK * (2 * DMODEL) * 2));
  // per-direction scratch region (reused afterwards for gate/FFN buffers)
  size_t sbase = off;
  bf16*  xz  = (bf16*)(ws + alloc((size_t)NTOK * 1024 * 2));
  bf16*  xc  = (bf16*)(ws + alloc((size_t)NTOK * DIN * 2));
  bf16*  dtb = (bf16*)(ws + alloc((size_t)NTOK * DIN * 2));
  float* bcb = (float*)(ws + alloc((size_t)NTOK * 32 * 4));
  bf16*  yg  = (bf16*)(ws + alloc((size_t)NTOK * DIN * 2));
  // post-phase aliases inside the scratch region (lifetimes disjoint):
  float* gate = (float*)(ws + sbase);
  bf16*  gmb  = (bf16*)(ws + sbase + (size_t)NTOK * DMODEL * 4);
  bf16*  h1   = (bf16*)(ws + sbase + (size_t)NTOK * DMODEL * 4 + (size_t)NTOK * DMODEL * 2);
  float* feb  = (float*)(ws + sbase + (size_t)NTOK * DMODEL * 4 + (size_t)NTOK * DMODEL * 4);

  auto cvt = [&](const float* in, bf16* o, int n) {
    cvt_f32_bf16<<<(n + 255) / 256, 256, 0, stream>>>(in, o, n);
  };

  // ---- prep: bf16 weight conversions + folded dt weights ----
  cvt(x, xb, NTOK * DMODEL);
  for (int d = 0; d < 2; ++d) {
    cvt(mp[d].Win,  win16[d],  1024 * 256);
    cvt(mp[d].Wx,   wx16[d],   48 * 512);
    cvt(mp[d].Wout, wout16[d], 256 * 512);
    wdt_eff_kernel<<<(512 * 512) / 256, 256, 0, stream>>>(mp[d].Wdt, mp[d].Wx, wdte16[d]);
  }
  cvt(gate_W, gw16, 256 * 512);
  cvt(fe_W1,  w116, 256 * 256);
  cvt(fe_W2,  w216, 256 * 256);

  // ---- both directions ----
  const long long convN = (long long)NTOK * DIN;
  for (int d = 0; d < 2; ++d) {
    const bool rev = (d == 1);
    // 1) xz = x @ Win^T    (65536x256 @ 256x1024)
    gemm_wmma<256, 4, ACT_NONE, false, true, false>
        <<<dim3(NTOK / 64, 1024 / 64), 128, 0, stream>>>(
            xb, win16[d], nullptr, xz, 1024, nullptr, 0, 0);
    // 2) depthwise conv + silu
    if (rev) conv_silu_kernel<true><<<(int)((convN + 255) / 256), 256, 0, stream>>>(xz, mp[d].convw, mp[d].convb, xc);
    else     conv_silu_kernel<false><<<(int)((convN + 255) / 256), 256, 0, stream>>>(xz, mp[d].convw, mp[d].convb, xc);
    // 3) dt = softplus(xc @ Wdt_eff^T + bdt)   (K=512, N=512)
    gemm_wmma<512, 4, ACT_SOFTPLUS, true, true, false>
        <<<dim3(NTOK / 64, 512 / 64), 128, 0, stream>>>(
            xc, wdte16[d], mp[d].bdt, dtb, 512, nullptr, 0, 0);
    // 4) [B|C] = xc @ Wx[16:48]^T              (K=512, N=32, fp32 out)
    gemm_wmma<512, 2, ACT_NONE, false, false, false>
        <<<dim3(NTOK / 64, 1), 128, 0, stream>>>(
            xc, wx16[d] + 16 * 512, nullptr, bcb, 32, nullptr, 0, 0);
    // 5) selective scan (+ skip u*D, + silu(z) gate fused)
    if (rev) scan_kernel<true><<<NGRAPH * 2, 256, 0, stream>>>(xc, dtb, bcb, xz, mp[d].Alog, mp[d].D, yg);
    else     scan_kernel<false><<<NGRAPH * 2, 256, 0, stream>>>(xc, dtb, bcb, xz, mp[d].Alog, mp[d].D, yg);
    // 6) dir output = yg @ Wout^T (fp32 for blend; bf16 copy into [f|b] concat)
    gemm_wmma<512, 4, ACT_NONE, false, false, true>
        <<<dim3(NTOK / 64, 256 / 64), 128, 0, stream>>>(
            yg, wout16[d], nullptr, rev ? b_out : f_out, 256,
            fb16, 512, rev ? 256 : 0);
  }

  // ---- gate, blend + LN1, FFN, residual + LN2 ----
  gemm_wmma<512, 4, ACT_SIGMOID, true, false, false>
      <<<dim3(NTOK / 64, 256 / 64), 128, 0, stream>>>(
          fb16, gw16, gate_b, gate, 256, nullptr, 0, 0);
  combine_ln1_kernel<<<NTOK, 256, 0, stream>>>(gate, f_out, b_out, ln1_g, ln1_b, gmb);
  gemm_wmma<256, 4, ACT_RELU, true, true, false>
      <<<dim3(NTOK / 64, 256 / 64), 128, 0, stream>>>(
          gmb, w116, fe_b1, h1, 256, nullptr, 0, 0);
  gemm_wmma<256, 4, ACT_NONE, true, false, false>
      <<<dim3(NTOK / 64, 256 / 64), 128, 0, stream>>>(
          h1, w216, fe_b2, feb, 256, nullptr, 0, 0);
  resid_ln2_kernel<<<NTOK, 256, 0, stream>>>(feb, x, ln2_g, ln2_b, outp);
}
